// GPT_62491774157469
// MI455X (gfx1250) — compile-verified, matrix operand order checked by reading, not run on hardware
//
#include <hip/hip_runtime.h>
#include <math.h>

// ---------------- problem constants ----------------
static constexpr int BB  = 2;
static constexpr int TT  = 1024;
static constexpr int DD  = 768;
static constexpr int HH  = 12;
static constexpr int DKK = 64;
static constexpr int FFF = 3072;
static constexpr int VV  = 50257;
static constexpr int LL  = 12;
static constexpr int MM  = BB * TT;   // 2048 token rows

typedef __attribute__((ext_vector_type(16))) _Float16 v16h;
typedef __attribute__((ext_vector_type(8)))  float    v8f;

// ---------------- WMMA wrapper ----------------
__device__ __forceinline__ v8f wmma_f16(v16h a, v16h b, v8f c) {
  return __builtin_amdgcn_wmma_f32_16x16x32_f16(
      false, a, false, b, (short)0, c, false, false);
}

// ---------------- async global->LDS staging (CDNA5, ASYNCcnt) ----------------
// One 128-bit DMA per lane; inst offset applies to BOTH lds and global address.
__device__ __forceinline__ void async_b128(unsigned lds, const void* gaddr) {
  asm volatile("global_load_async_to_lds_b128 %0, %1, off"
               :: "v"(lds), "v"((unsigned long long)gaddr) : "memory");
}
__device__ __forceinline__ void async_b128_off16(unsigned lds, const void* gaddr) {
  asm volatile("global_load_async_to_lds_b128 %0, %1, off offset:16"
               :: "v"(lds), "v"((unsigned long long)gaddr) : "memory");
}
__device__ __forceinline__ void async_b128_off32(unsigned lds, const void* gaddr) {
  asm volatile("global_load_async_to_lds_b128 %0, %1, off offset:32"
               :: "v"(lds), "v"((unsigned long long)gaddr) : "memory");
}
__device__ __forceinline__ void async_b128_off48(unsigned lds, const void* gaddr) {
  asm volatile("global_load_async_to_lds_b128 %0, %1, off offset:48"
               :: "v"(lds), "v"((unsigned long long)gaddr) : "memory");
}
__device__ __forceinline__ void wait_async0() {
  asm volatile("s_wait_asynccnt 0x0" ::: "memory");
}

// ---------------- LDS fragment loaders (wave32 layouts) ----------------
__device__ __forceinline__ v16h lds_a(const _Float16* base, int ld, int koff) {
  const int lane = threadIdx.x & 31;
  const int m = lane & 15, h = lane >> 4;
  v16h a;
#pragma unroll
  for (int v = 0; v < 8; ++v) {
    int k = koff + ((v >> 2) << 4) + ((v & 3) << 1) + (h << 3);
    a[2 * v]     = base[m * ld + k];
    a[2 * v + 1] = base[m * ld + k + 1];
  }
  return a;
}

__device__ __forceinline__ v16h lds_b(const _Float16* base, int ld, int col0, int koff) {
  const int lane = threadIdx.x & 31;
  const int n = lane & 15, h = lane >> 4;
  v16h b;
#pragma unroll
  for (int v = 0; v < 8; ++v) {
    int k = koff + ((v >> 2) << 4) + ((v & 3) << 1) + (h << 3);
    b[2 * v]     = base[k * ld + col0 + n];
    b[2 * v + 1] = base[(k + 1) * ld + col0 + n];
  }
  return b;
}

__device__ __forceinline__ v16h lds_bt(const _Float16* base, int ld, int row0, int koff) {
  const int lane = threadIdx.x & 31;
  const int n = lane & 15, h = lane >> 4;
  v16h b;
#pragma unroll
  for (int v = 0; v < 8; ++v) {
    int k = koff + ((v >> 2) << 4) + ((v & 3) << 1) + (h << 3);
    b[2 * v]     = base[(row0 + n) * ld + k];
    b[2 * v + 1] = base[(row0 + n) * ld + k + 1];
  }
  return b;
}

// ---------------- embedding ----------------
__global__ __launch_bounds__(256) void k_embed(const int* __restrict__ ids,
                                               const float* __restrict__ tok,
                                               const float* __restrict__ pos,
                                               float* __restrict__ h) {
  const int row = blockIdx.x;
  const int t   = row % TT;
  const int id  = ids[row];
  const float* te = tok + (size_t)id * DD;
  const float* pe = pos + (size_t)t * DD;
  float* o = h + (size_t)row * DD;
  for (int i = threadIdx.x; i < DD; i += blockDim.x) o[i] = te[i] + pe[i];
}

// ---------------- layernorm (fp32 in -> f16 out) ----------------
__global__ __launch_bounds__(256) void k_ln(const float* __restrict__ X,
                                            const float* __restrict__ g,
                                            const float* __restrict__ bta,
                                            _Float16* __restrict__ Y) {
  const int row = blockIdx.x;
  const float* x = X + (size_t)row * DD;
  float s = 0.f, ss = 0.f;
  for (int i = threadIdx.x; i < DD; i += 256) {
    float v = x[i]; s += v; ss += v * v;
  }
  __shared__ float rs[256], rss[256];
  const int tid = threadIdx.x;
  rs[tid] = s; rss[tid] = ss;
  __syncthreads();
  for (int off = 128; off > 0; off >>= 1) {
    if (tid < off) { rs[tid] += rs[tid + off]; rss[tid] += rss[tid + off]; }
    __syncthreads();
  }
  const float mean = rs[0] * (1.0f / DD);
  const float var  = rss[0] * (1.0f / DD) - mean * mean;
  const float rstd = rsqrtf(var + 1e-5f);
  _Float16* y = Y + (size_t)row * DD;
  for (int i = threadIdx.x; i < DD; i += 256)
    y[i] = (_Float16)((x[i] - mean) * rstd * g[i] + bta[i]);
}

// ---------------- double-buffered tiled WMMA GEMM ----------------
// C[M,N] = act( A[M,K](f16) @ B + bias ) (+ resid), B fp32 converted in-flight:
//   BT=false : B stored [K,N] row-major (weights)
//   BT=true  : B stored [N,K] row-major (tied lm_head tok_emb [V,D]); N bounds-checked,
//              fp32 output stored non-temporally (keep tok_emb resident in L2 instead).
template <bool BT, bool GELU, bool RESID, bool OUTF16>
__global__ __launch_bounds__(256) void k_gemm(const _Float16* __restrict__ A,
                                              const float* __restrict__ Bw,
                                              const float* __restrict__ bias,
                                              const float* __restrict__ resid,
                                              float* __restrict__ Cf,
                                              _Float16* __restrict__ Ch,
                                              int M, int N, int K) {
  constexpr int BM = 128, BN = 128, BK = 32, BLD = 136;
  __shared__ __align__(16) _Float16 As[2][BM * BK];
  __shared__ __align__(16) _Float16 Bs[2][BK * BLD];

  const int tid  = threadIdx.x;
  const int lane = tid & 31;
  const int w    = tid >> 5;
  const int wm   = w >> 1;         // 0..3  -> 32-row strip
  const int wn   = w & 1;          // 0..1  -> 64-col strip
  const int m0   = blockIdx.y * BM;
  const int n0   = blockIdx.x * BN;

  // per-thread staging coordinates
  const int rowA = tid >> 1, khA = (tid & 1) * 16;     // A: 16 halves/thread
  const int kkB  = tid >> 3, nbB = (tid & 7) * 16;     // B (!BT): 16 floats/thread
  const int nnB  = tid >> 1, khB = (tid & 1) * 16;     // B (BT)

  v8f acc[2][4];
  const v8f vzero = {0.f, 0.f, 0.f, 0.f, 0.f, 0.f, 0.f, 0.f};
#pragma unroll
  for (int i = 0; i < 2; ++i)
#pragma unroll
    for (int j = 0; j < 4; ++j) acc[i][j] = vzero;

  uint4 aR0, aR1;
  float bR[16];

  auto load_tile = [&](int k0) {
    const uint4* sa = reinterpret_cast<const uint4*>(A + (size_t)(m0 + rowA) * K + k0 + khA);
    aR0 = sa[0]; aR1 = sa[1];
    if (!BT) {
      const float* sb = Bw + (size_t)(k0 + kkB) * N + n0 + nbB;
#pragma unroll
      for (int j = 0; j < 16; ++j) bR[j] = sb[j];
    } else {
      if (n0 + nnB < N) {
        const float* sb = Bw + (size_t)(n0 + nnB) * K + k0 + khB;
#pragma unroll
        for (int j = 0; j < 16; ++j) bR[j] = sb[j];
      } else {
#pragma unroll
        for (int j = 0; j < 16; ++j) bR[j] = 0.0f;
      }
    }
  };
  auto store_tile = [&](int buf) {
    uint4* da = reinterpret_cast<uint4*>(&As[buf][rowA * BK + khA]);
    da[0] = aR0; da[1] = aR1;
    if (!BT) {
#pragma unroll
      for (int j = 0; j < 16; ++j) Bs[buf][kkB * BLD + nbB + j] = (_Float16)bR[j];
    } else {
#pragma unroll
      for (int j = 0; j < 16; ++j) Bs[buf][(khB + j) * BLD + nnB] = (_Float16)bR[j];
    }
  };

  // prologue
  load_tile(0);
  store_tile(0);
  __syncthreads();

  const int ntiles = K / BK;
  for (int i = 0; i < ntiles; ++i) {
    const int buf = i & 1;
    if (i + 1 < ntiles) load_tile((i + 1) * BK);   // globals in flight during WMMA

    const v16h a0 = lds_a(&As[buf][(wm * 32) * BK], BK, 0);
    const v16h a1 = lds_a(&As[buf][(wm * 32 + 16) * BK], BK, 0);
#pragma unroll
    for (int ni = 0; ni < 4; ++ni) {
      const v16h b = lds_b(&Bs[buf][0], BLD, wn * 64 + ni * 16, 0);
      acc[0][ni] = wmma_f16(a0, b, acc[0][ni]);
      acc[1][ni] = wmma_f16(a1, b, acc[1][ni]);
    }

    if (i + 1 < ntiles) store_tile(buf ^ 1);
    __syncthreads();
  }

  // --- epilogue: bias / GELU / residual / store ---
  const int nl = lane & 15, mh = lane >> 4;
#pragma unroll
  for (int mi = 0; mi < 2; ++mi) {
#pragma unroll
    for (int ni = 0; ni < 4; ++ni) {
      const int col = n0 + wn * 64 + ni * 16 + nl;
      const bool colok = (!BT) || (col < N);
      const float bv = (bias != nullptr && colok) ? bias[col] : 0.0f;
#pragma unroll
      for (int r = 0; r < 8; ++r) {
        const int row = m0 + wm * 32 + mi * 16 + mh * 8 + r;
        float v = acc[mi][ni][r] + bv;
        if (GELU) v = 0.5f * v * (1.0f + erff(v * 0.70710678118654752f));
        if (colok) {
          const size_t idx = (size_t)row * N + col;
          if (RESID) v += resid[idx];
          if (OUTF16) {
            Ch[idx] = (_Float16)v;
          } else if (BT) {
            __builtin_nontemporal_store(v, &Cf[idx]);  // logits: don't pollute L2
          } else {
            Cf[idx] = v;
          }
        }
      }
    }
  }
}

// ---------------- flash attention (causal, online softmax) ----------------
// q/k/v/ctx are f16 [B*T, D] with D index = head*64 + dk.
__global__ __launch_bounds__(256) void k_attn(const _Float16* __restrict__ Q,
                                              const _Float16* __restrict__ Kx,
                                              const _Float16* __restrict__ Vx,
                                              _Float16* __restrict__ O) {
  constexpr int QT = 128, KT = 64, LDQ = 72;
  __shared__ __align__(16) _Float16 Qs[QT * LDQ];
  __shared__ __align__(16) _Float16 Ks[KT * LDQ];
  __shared__ __align__(16) _Float16 Vs[KT * LDQ];
  __shared__ __align__(16) _Float16 Ps[QT * LDQ];

  const int tid  = threadIdx.x;
  const int lane = tid & 31;
  const int w    = tid >> 5;                  // 8 waves x 16 query rows
  const int q0   = blockIdx.x * QT;
  const int head = blockIdx.y;
  const int b    = blockIdx.z;
  const size_t base = (size_t)b * TT * DD + (size_t)head * DKK;
  const _Float16* Qb = Q  + base;
  const _Float16* Kb = Kx + base;
  const _Float16* Vb = Vx + base;

  // stage Q tile (128 x 64) via async DMA: 64B per thread = 4 x b128
  {
    const int row = tid >> 1, c0 = (tid & 1) * 32;
    const unsigned ldsq = (unsigned)(size_t)&Qs[row * LDQ + c0];
    const void* gq = (const void*)(Qb + (size_t)(q0 + row) * DD + c0);
    async_b128(ldsq, gq);
    async_b128_off16(ldsq, gq);
    async_b128_off32(ldsq, gq);
    async_b128_off48(ldsq, gq);
    wait_async0();
  }
  __syncthreads();
  v16h qf[2];
  qf[0] = lds_a(&Qs[(w * 16) * LDQ], LDQ, 0);
  qf[1] = lds_a(&Qs[(w * 16) * LDQ], LDQ, 32);

  const v8f vzero = {0.f, 0.f, 0.f, 0.f, 0.f, 0.f, 0.f, 0.f};
  v8f acc[4];
#pragma unroll
  for (int ni = 0; ni < 4; ++ni) acc[ni] = vzero;
  float rowmax[8], rowsum[8];
#pragma unroll
  for (int r = 0; r < 8; ++r) { rowmax[r] = -1e30f; rowsum[r] = 0.f; }

  const int ncol  = lane & 15;
  const int mrow0 = q0 + w * 16 + (lane >> 4) * 8;
  const int nkt   = q0 / KT + 2;              // causal: keys up to q0+127

  for (int kt = 0; kt < nkt; ++kt) {
    const int kb = kt * KT;
    __syncthreads();
    // stage K/V tiles (64 x 64) via async DMA: 32B per thread each = 2 x b128
    {
      const int row = tid >> 2, c0 = (tid & 3) * 16;
      const unsigned ldsk = (unsigned)(size_t)&Ks[row * LDQ + c0];
      const unsigned ldsv = (unsigned)(size_t)&Vs[row * LDQ + c0];
      const void* gk = (const void*)(Kb + (size_t)(kb + row) * DD + c0);
      const void* gv = (const void*)(Vb + (size_t)(kb + row) * DD + c0);
      async_b128(ldsk, gk);
      async_b128_off16(ldsk, gk);
      async_b128(ldsv, gv);
      async_b128_off16(ldsv, gv);
      wait_async0();
    }
    __syncthreads();

    // S = Q @ K^T
    v8f s[4];
#pragma unroll
    for (int ni = 0; ni < 4; ++ni) {
      s[ni] = vzero;
#pragma unroll
      for (int c = 0; c < 2; ++c) {
        const v16h kf = lds_bt(Ks, LDQ, ni * 16, c * 32);
        s[ni] = wmma_f16(qf[c], kf, s[ni]);
      }
    }
    // scale + causal mask
#pragma unroll
    for (int ni = 0; ni < 4; ++ni)
#pragma unroll
      for (int r = 0; r < 8; ++r) {
        float v = s[ni][r] * 0.125f;          // 1/sqrt(64)
        const int key = kb + ni * 16 + ncol;
        if (key > mrow0 + r) v = -1e30f;
        s[ni][r] = v;
      }
    // online softmax per owned row (16-lane half reductions)
#pragma unroll
    for (int r = 0; r < 8; ++r) {
      float tmax = s[0][r];
#pragma unroll
      for (int ni = 1; ni < 4; ++ni) tmax = fmaxf(tmax, s[ni][r]);
#pragma unroll
      for (int msk = 1; msk < 16; msk <<= 1) tmax = fmaxf(tmax, __shfl_xor(tmax, msk, 16));
      const float newm  = fmaxf(rowmax[r], tmax);
      const float alpha = expf(rowmax[r] - newm);
      float ls = 0.f;
#pragma unroll
      for (int ni = 0; ni < 4; ++ni) {
        const float p = expf(s[ni][r] - newm);
        s[ni][r] = p; ls += p;
      }
#pragma unroll
      for (int msk = 1; msk < 16; msk <<= 1) ls += __shfl_xor(ls, msk, 16);
      rowsum[r] = rowsum[r] * alpha + ls;
      rowmax[r] = newm;
#pragma unroll
      for (int ni = 0; ni < 4; ++ni) acc[ni][r] *= alpha;
    }
    // restage P (C-layout f32 -> A-layout f16 via per-wave LDS region)
#pragma unroll
    for (int ni = 0; ni < 4; ++ni)
#pragma unroll
      for (int r = 0; r < 8; ++r)
        Ps[(w * 16 + (lane >> 4) * 8 + r) * LDQ + ni * 16 + ncol] = (_Float16)s[ni][r];
    __syncthreads();

    v16h pf[2];
    pf[0] = lds_a(&Ps[(w * 16) * LDQ], LDQ, 0);
    pf[1] = lds_a(&Ps[(w * 16) * LDQ], LDQ, 32);
#pragma unroll
    for (int ni = 0; ni < 4; ++ni)
#pragma unroll
      for (int c = 0; c < 2; ++c) {
        const v16h vf = lds_b(Vs, LDQ, ni * 16, c * 32);
        acc[ni] = wmma_f16(pf[c], vf, acc[ni]);
      }
  }

  // normalize + write ctx (f16)
#pragma unroll
  for (int ni = 0; ni < 4; ++ni)
#pragma unroll
    for (int r = 0; r < 8; ++r) {
      const int row = q0 + w * 16 + (lane >> 4) * 8 + r;
      const float inv = 1.0f / rowsum[r];
      O[(size_t)((size_t)b * TT + row) * DD + head * DKK + ni * 16 + ncol] =
          (_Float16)(acc[ni][r] * inv);
    }
}

// ---------------- host orchestration ----------------
extern "C" void kernel_launch(void* const* d_in, const int* in_sizes, int n_in,
                              void* d_out, int out_size, void* d_ws, size_t ws_size,
                              hipStream_t stream) {
  (void)in_sizes; (void)n_in; (void)out_size; (void)ws_size;
  const int*   ids  = (const int*)d_in[0];
  const float* tok  = (const float*)d_in[1];
  const float* pos  = (const float*)d_in[2];
  const float* Wq   = (const float*)d_in[3];
  const float* bq   = (const float*)d_in[4];
  const float* Wk   = (const float*)d_in[5];
  const float* bk   = (const float*)d_in[6];
  const float* Wv   = (const float*)d_in[7];
  const float* bv   = (const float*)d_in[8];
  const float* Wo   = (const float*)d_in[9];
  const float* bo   = (const float*)d_in[10];
  const float* ln1g = (const float*)d_in[11];
  const float* ln1b = (const float*)d_in[12];
  const float* ln2g = (const float*)d_in[13];
  const float* ln2b = (const float*)d_in[14];
  const float* W1   = (const float*)d_in[15];
  const float* b1   = (const float*)d_in[16];
  const float* W2   = (const float*)d_in[17];
  const float* b2   = (const float*)d_in[18];
  const float* lnfg = (const float*)d_in[19];
  const float* lnfb = (const float*)d_in[20];

  char* ws = (char*)d_ws;
  const size_t szH = (size_t)MM * DD * sizeof(float);
  const size_t szA = (size_t)MM * DD * sizeof(_Float16);
  float*    h    = (float*)ws;
  _Float16* x2h  = (_Float16*)(ws + szH);
  _Float16* qh   = (_Float16*)(ws + szH + szA);
  _Float16* kh   = (_Float16*)(ws + szH + 2 * szA);
  _Float16* vh   = (_Float16*)(ws + szH + 3 * szA);
  _Float16* ctxh = (_Float16*)(ws + szH + 4 * szA);
  _Float16* ffh  = (_Float16*)(ws + szH + 5 * szA);

  const dim3 blk(256);
  const dim3 gDD((DD + 127) / 128, MM / 128);
  const dim3 gFF((FFF + 127) / 128, MM / 128);
  const dim3 gV((VV + 127) / 128, MM / 128);
  const dim3 gAttn(TT / 128, HH, BB);

  k_embed<<<MM, blk, 0, stream>>>(ids, tok, pos, h);

  for (int l = 0; l < LL; ++l) {
    const size_t oDD = (size_t)l * DD * DD;
    const size_t oDF = (size_t)l * DD * FFF;
    k_ln<<<MM, blk, 0, stream>>>(h, ln1g + (size_t)l * DD, ln1b + (size_t)l * DD, x2h);
    k_gemm<false, false, false, true><<<gDD, blk, 0, stream>>>(
        x2h, Wq + oDD, bq + (size_t)l * DD, nullptr, nullptr, qh, MM, DD, DD);
    k_gemm<false, false, false, true><<<gDD, blk, 0, stream>>>(
        x2h, Wk + oDD, bk + (size_t)l * DD, nullptr, nullptr, kh, MM, DD, DD);
    k_gemm<false, false, false, true><<<gDD, blk, 0, stream>>>(
        x2h, Wv + oDD, bv + (size_t)l * DD, nullptr, nullptr, vh, MM, DD, DD);
    k_attn<<<gAttn, blk, 0, stream>>>(qh, kh, vh, ctxh);
    k_gemm<false, false, true, false><<<gDD, blk, 0, stream>>>(
        ctxh, Wo + oDD, bo + (size_t)l * DD, h, h, nullptr, MM, DD, DD);
    k_ln<<<MM, blk, 0, stream>>>(h, ln2g + (size_t)l * DD, ln2b + (size_t)l * DD, x2h);
    k_gemm<false, true, false, true><<<gFF, blk, 0, stream>>>(
        x2h, W1 + oDF, b1 + (size_t)l * FFF, nullptr, nullptr, ffh, MM, FFF, DD);
    k_gemm<false, false, true, false><<<gDD, blk, 0, stream>>>(
        ffh, W2 + oDF, b2 + (size_t)l * DD, h, h, nullptr, MM, DD, FFF);
  }

  k_ln<<<MM, blk, 0, stream>>>(h, lnfg, lnfb, x2h);
  k_gemm<true, false, false, false><<<gV, blk, 0, stream>>>(
      x2h, tok, nullptr, nullptr, (float*)d_out, nullptr, MM, VV, DD);
}
